// HRP_3994319585787
// MI455X (gfx1250) — matrix operand order repaired and checked by reading, not compiled, not run on hardware
//
#include <hip/hip_runtime.h>

typedef __attribute__((ext_vector_type(16))) _Float16 v16h;
typedef __attribute__((ext_vector_type(8)))  _Float16 v8h;
typedef __attribute__((ext_vector_type(8)))  float    v8f;

// ---------------------------------------------------------------------------
// WMMA helpers (CDNA5 gfx1250, wave32, 16x16x32 f16 -> f32)
// A-fragment (16-bit A 16x32, ISA 7.12.2): lane holds row m=lane&15; lanes 0-15
// hold K groups {0..7, 16..23}, lanes 16-31 hold {8..15, 24..31} -> two
// contiguous 8-half groups => two ds_load_b128 from LDS.
// B-fragment: prepped in workspace so each lane's 16 halves are contiguous.
// ---------------------------------------------------------------------------
__device__ __forceinline__ v16h a_frag_lds(const _Float16* base, int stride) {
  int lane = threadIdx.x & 31;
  int m  = lane & 15;
  int kg = (lane >> 4) * 8;
  const _Float16* p = base + m * stride + kg;
  v8h lo = *(const v8h*)(p);
  v8h hi = *(const v8h*)(p + 16);
  v16h r;
#pragma unroll
  for (int i = 0; i < 8; ++i) { r[i] = lo[i]; r[i + 8] = hi[i]; }
  return r;
}

__device__ __forceinline__ v16h b_frag_ws(const _Float16* tiles, int tile_idx) {
  int lane = threadIdx.x & 31;
  return *(const v16h*)(tiles + (size_t)tile_idx * 512 + lane * 16);
}

__device__ __forceinline__ v8f wmma_f16(v16h a, v16h b, v8f c) {
  return __builtin_amdgcn_wmma_f32_16x16x32_f16(false, a, false, b, (short)0, c,
                                                false, false);
}

__device__ __forceinline__ float sigm(float x) {
  return 1.f / (1.f + __expf(-x));
}

// GEMM: A[16,K] (LDS, f16, row stride `astride`) x B (prepped tiles) -> relu/bias
// -> out[16, NT*16] (LDS f16, row stride `ostride`).
template <int NT, int KS>
__device__ __forceinline__ void gemm_bias_relu(const _Float16* A, int astride,
                                               const _Float16* Bt,
                                               const float* __restrict__ bias,
                                               _Float16* out, int ostride,
                                               bool relu) {
  int lane = threadIdx.x & 31;
  int n = lane & 15;
  int mb = (lane >> 4) * 8;
  for (int nt = 0; nt < NT; ++nt) {
    v8f acc = {};
#pragma unroll
    for (int ks = 0; ks < KS; ++ks) {
      v16h a = a_frag_lds(A + ks * 32, astride);
      acc = wmma_f16(a, b_frag_ws(Bt, nt * KS + ks), acc);
    }
    float bb = bias[nt * 16 + n];
#pragma unroll
    for (int i = 0; i < 8; ++i) {
      float v = acc[i] + bb;
      if (relu) v = fmaxf(v, 0.f);
      out[(mb + i) * ostride + nt * 16 + n] = (_Float16)v;
    }
  }
}

// ---------------------------------------------------------------------------
// Weight prep: w is [N, Kreal] row-major f32. Emit B = w^T as f16 fragment
// tiles: tile = ntile*ksteps + kstep; within a tile lane's 16 halves are
// contiguous. K zero-padded to ksteps*32.
// ---------------------------------------------------------------------------
__global__ void __launch_bounds__(32)
prep_btiles(const float* __restrict__ w, _Float16* __restrict__ dst, int Kreal,
            int ksteps) {
  int tile = blockIdx.x;
  int ntile = tile / ksteps;
  int kstep = tile % ksteps;
  int lane = threadIdx.x;
  int n = ntile * 16 + (lane & 15);
  int kbase = kstep * 32 + (lane >> 4) * 8;
  _Float16* o = dst + (size_t)tile * 512 + lane * 16;
#pragma unroll
  for (int i = 0; i < 16; ++i) {
    int k = kbase + i + (i >= 8 ? 8 : 0);
    float v = (k < Kreal) ? w[(size_t)n * Kreal + k] : 0.f;
    o[i] = (_Float16)v;
  }
}

// ---------------------------------------------------------------------------
// Actor: 16 rows per 1-wave block over 102400 rows (row = z*1024 + b).
// GRU(6->64) then MLP 64->512->128->1 per timestep; write sa_flat[row*2+t].
// ---------------------------------------------------------------------------
__global__ void __launch_bounds__(32)
actor_kernel(const float* __restrict__ x, const float* __restrict__ wih,
             const float* __restrict__ bih, const float* __restrict__ bhh,
             const _Float16* __restrict__ whh_t,
             const _Float16* __restrict__ fc1_t, const float* __restrict__ fc1_b,
             const _Float16* __restrict__ fc2_t, const float* __restrict__ fc2_b,
             const float* __restrict__ fc3_w, const float* __restrict__ fc3_b,
             float* __restrict__ sa) {
  __shared__ float xs[16][2][6];
  __shared__ float wihs[192 * 6];
  __shared__ float bihs[192], bhhs[192];
  __shared__ _Float16 hbuf[16 * 64];
  __shared__ _Float16 act1[16 * 512];
  __shared__ _Float16 act2[16 * 128];

  int lane = threadIdx.x;
  int row0 = blockIdx.x * 16;

  for (int i = lane; i < 192 * 6; i += 32) wihs[i] = wih[i];
  for (int i = lane; i < 192; i += 32) { bihs[i] = bih[i]; bhhs[i] = bhh[i]; }
  for (int i = lane; i < 16 * 2 * 6; i += 32) {
    int m = i / 12, rem = i % 12, t = rem / 6, s = rem % 6;
    int r = row0 + m;
    int z = r >> 10, b = r & 1023;                       // row = z*1024 + b
    xs[m][t][s] = x[((size_t)b * 2 + t) * 600 + z * 6 + s];
  }
  for (int i = lane; i < 16 * 64; i += 32) hbuf[i] = (_Float16)0.f;
  __syncthreads();

  int n = lane & 15;
  int mb = (lane >> 4) * 8;

  for (int t = 0; t < 2; ++t) {
    // gh = h @ Whh^T : [16,64] x [64,192], 12 n-tiles x 2 k-steps
    v16h a0 = a_frag_lds(hbuf + 0, 64);
    v16h a1 = a_frag_lds(hbuf + 32, 64);
    v8f gh[12];
#pragma unroll
    for (int nt = 0; nt < 12; ++nt) {
      v8f acc = {};
      acc = wmma_f16(a0, b_frag_ws(whh_t, nt * 2 + 0), acc);
      acc = wmma_f16(a1, b_frag_ws(whh_t, nt * 2 + 1), acc);
      gh[nt] = acc;
    }
    __syncthreads();
    // GRU gate update (gi computed in VALU; K=6 too thin for WMMA).
#pragma unroll
    for (int ct = 0; ct < 4; ++ct) {
      int j = ct * 16 + n;  // hidden unit
#pragma unroll
      for (int i = 0; i < 8; ++i) {
        int m = mb + i;
        float xr = bihs[j], xz = bihs[64 + j], xn = bihs[128 + j];
#pragma unroll
        for (int s = 0; s < 6; ++s) {
          float xv = xs[m][t][s];
          xr += xv * wihs[j * 6 + s];
          xz += xv * wihs[(64 + j) * 6 + s];
          xn += xv * wihs[(128 + j) * 6 + s];
        }
        float hr = gh[ct][i] + bhhs[j];
        float hz = gh[4 + ct][i] + bhhs[64 + j];
        float hn = gh[8 + ct][i] + bhhs[128 + j];
        float rg = sigm(xr + hr);
        float zg = sigm(xz + hz);
        float ng = tanhf(xn + rg * hn);
        float ho = (float)hbuf[m * 64 + j];
        hbuf[m * 64 + j] = (_Float16)((1.f - zg) * ng + zg * ho);
      }
    }
    __syncthreads();
    // fc1: [16,64] x [64,512]
    gemm_bias_relu<32, 2>(hbuf, 64, fc1_t, fc1_b, act1, 512, true);
    __syncthreads();
    // fc2: [16,512] x [512,128]
    gemm_bias_relu<8, 16>(act1, 512, fc2_t, fc2_b, act2, 128, true);
    __syncthreads();
    // fc3: 128 -> 1 (per-row dot, lanes 0..15)
    if (lane < 16) {
      float s = fc3_b[0];
      for (int k = 0; k < 128; ++k) s += (float)act2[lane * 128 + k] * fc3_w[k];
      sa[(size_t)(row0 + lane) * 2 + t] = s;  // sa_flat[z*2048 + b*2 + t]
    }
    __syncthreads();
  }
}

// p[:, -1] gather: out[n*100+z] = sa_flat[n*200 + 100 + z]
__global__ void gather_p(const float* __restrict__ sa, float* __restrict__ out) {
  int i = blockIdx.x * blockDim.x + threadIdx.x;
  if (i < 102400) {
    int nrow = i / 100, z = i % 100;
    out[i] = sa[(size_t)nrow * 200 + 100 + z];
  }
}

// ---------------------------------------------------------------------------
// Critic: 16 batch rows per 1-wave block (64 blocks). Builds xi (zones
// {99,89,0,0,98}, 35 feats zero-padded to 64), LocalizedModule on t=1,
// GRU(35->32) + MLP 32->512->256->1; q = f + qz.
// ---------------------------------------------------------------------------
__global__ void __launch_bounds__(32)
critic_kernel(const float* __restrict__ x, const float* __restrict__ sa,
              const _Float16* __restrict__ cwih_t,
              const _Float16* __restrict__ cwhh_t,
              const float* __restrict__ cbih, const float* __restrict__ cbhh,
              const _Float16* __restrict__ cfc1_t, const float* __restrict__ cfc1_b,
              const _Float16* __restrict__ cfc2_t, const float* __restrict__ cfc2_b,
              const float* __restrict__ cfc3_w, const float* __restrict__ cfc3_b,
              const _Float16* __restrict__ lfc1_t, const float* __restrict__ lfc1_b,
              const _Float16* __restrict__ lfc2_t, const float* __restrict__ lfc2_b,
              const float* __restrict__ lfc3_w, const float* __restrict__ lfc3_b,
              float* __restrict__ qout) {
  __shared__ _Float16 xib[2][16 * 64];
  __shared__ _Float16 hb[16 * 32];
  __shared__ _Float16 act1[16 * 512];
  __shared__ _Float16 act2[16 * 256];
  __shared__ float ftmp[16];

  int lane = threadIdx.x;
  int b0 = blockIdx.x * 16;
  int n = lane & 15;
  int mb = (lane >> 4) * 8;

  for (int i = lane; i < 2 * 16 * 64; i += 32) (&xib[0][0])[i] = (_Float16)0.f;
  for (int i = lane; i < 16 * 32; i += 32) hb[i] = (_Float16)0.f;
  __syncthreads();
  // 16 rows * 2 t * 5 groups * 7 features
  for (int i = lane; i < 16 * 2 * 5 * 7; i += 32) {
    int m = i / 70, rem = i % 70, t = rem / 35, c = rem % 35;
    int g = c / 7, s = c % 7;
    int z = (g == 0) ? 99 : (g == 1) ? 89 : (g == 4) ? 98 : -1;
    float v = 0.f;
    if (z >= 0) {
      int b = b0 + m;
      if (s < 6) v = x[((size_t)b * 2 + t) * 600 + z * 6 + s];
      else       v = sa[(size_t)b * 200 + t * 100 + z];  // action feature
    }
    xib[t][m * 64 + c] = (_Float16)v;
  }
  __syncthreads();

  // ---- LocalizedModule on xi[:, 1, :]: 35(pad64)->512->256->1 ----
  gemm_bias_relu<32, 2>(xib[1], 64, lfc1_t, lfc1_b, act1, 512, true);
  __syncthreads();
  gemm_bias_relu<16, 16>(act1, 512, lfc2_t, lfc2_b, act2, 256, true);
  __syncthreads();
  if (lane < 16) {
    float s = lfc3_b[0];
    for (int k = 0; k < 256; ++k) s += (float)act2[lane * 256 + k] * lfc3_w[k];
    ftmp[lane] = s;
  }
  __syncthreads();

  // ---- critic GRU(35->32) ----
  for (int t = 0; t < 2; ++t) {
    v16h xa0 = a_frag_lds(xib[t] + 0, 64);
    v16h xa1 = a_frag_lds(xib[t] + 32, 64);
    v16h ha = a_frag_lds(hb, 32);
    v8f gi[6], gh[6];
#pragma unroll
    for (int nt = 0; nt < 6; ++nt) {
      v8f acc = {};
      acc = wmma_f16(xa0, b_frag_ws(cwih_t, nt * 2 + 0), acc);
      acc = wmma_f16(xa1, b_frag_ws(cwih_t, nt * 2 + 1), acc);
      gi[nt] = acc;
      v8f acc2 = {};
      gh[nt] = wmma_f16(ha, b_frag_ws(cwhh_t, nt), acc2);
    }
    __syncthreads();
#pragma unroll
    for (int ct = 0; ct < 2; ++ct) {
      int j = ct * 16 + n;  // hidden unit (H=32)
#pragma unroll
      for (int i = 0; i < 8; ++i) {
        int m = mb + i;
        float ir = gi[ct][i] + cbih[j];
        float iz = gi[2 + ct][i] + cbih[32 + j];
        float in = gi[4 + ct][i] + cbih[64 + j];
        float hr = gh[ct][i] + cbhh[j];
        float hz = gh[2 + ct][i] + cbhh[32 + j];
        float hn = gh[4 + ct][i] + cbhh[64 + j];
        float rg = sigm(ir + hr);
        float zg = sigm(iz + hz);
        float ng = tanhf(in + rg * hn);
        float ho = (float)hb[m * 32 + j];
        hb[m * 32 + j] = (_Float16)((1.f - zg) * ng + zg * ho);
      }
    }
    __syncthreads();
  }

  // ---- SubCritic MLP on last hidden: 32->512->256->1 ----
  gemm_bias_relu<32, 1>(hb, 32, cfc1_t, cfc1_b, act1, 512, true);
  __syncthreads();
  gemm_bias_relu<16, 16>(act1, 512, cfc2_t, cfc2_b, act2, 256, true);
  __syncthreads();
  if (lane < 16) {
    float s = cfc3_b[0];
    for (int k = 0; k < 256; ++k) s += (float)act2[lane * 256 + k] * cfc3_w[k];
    qout[b0 + lane] = ftmp[lane] + s;
  }
}

// ---------------------------------------------------------------------------
extern "C" void kernel_launch(void* const* d_in, const int* in_sizes, int n_in,
                              void* d_out, int out_size, void* d_ws,
                              size_t ws_size, hipStream_t stream) {
  (void)in_sizes; (void)n_in; (void)out_size; (void)ws_size;
  const float* x      = (const float*)d_in[0];
  const float* a_wih  = (const float*)d_in[1];
  const float* a_whh  = (const float*)d_in[2];
  const float* a_bih  = (const float*)d_in[3];
  const float* a_bhh  = (const float*)d_in[4];
  const float* a_fc1w = (const float*)d_in[5];
  const float* a_fc1b = (const float*)d_in[6];
  const float* a_fc2w = (const float*)d_in[7];
  const float* a_fc2b = (const float*)d_in[8];
  const float* a_fc3w = (const float*)d_in[9];
  const float* a_fc3b = (const float*)d_in[10];
  const float* c_wih  = (const float*)d_in[11];
  const float* c_whh  = (const float*)d_in[12];
  const float* c_bih  = (const float*)d_in[13];
  const float* c_bhh  = (const float*)d_in[14];
  const float* c_fc1w = (const float*)d_in[15];
  const float* c_fc1b = (const float*)d_in[16];
  const float* c_fc2w = (const float*)d_in[17];
  const float* c_fc2b = (const float*)d_in[18];
  const float* c_fc3w = (const float*)d_in[19];
  const float* c_fc3b = (const float*)d_in[20];
  const float* l_fc1w = (const float*)d_in[21];
  const float* l_fc1b = (const float*)d_in[22];
  const float* l_fc2w = (const float*)d_in[23];
  const float* l_fc2b = (const float*)d_in[24];
  const float* l_fc3w = (const float*)d_in[25];
  const float* l_fc3b = (const float*)d_in[26];

  float* sa = (float*)d_ws;                              // 102400*2 f32
  _Float16* tb = (_Float16*)((char*)d_ws + 102400 * 2 * sizeof(float));
  _Float16* t_awhh = tb + (size_t)0 * 512;    // 24 tiles
  _Float16* t_afc1 = tb + (size_t)24 * 512;   // 64
  _Float16* t_afc2 = tb + (size_t)88 * 512;   // 128
  _Float16* t_cwih = tb + (size_t)216 * 512;  // 12
  _Float16* t_cwhh = tb + (size_t)228 * 512;  // 6
  _Float16* t_cfc1 = tb + (size_t)234 * 512;  // 32
  _Float16* t_cfc2 = tb + (size_t)266 * 512;  // 256
  _Float16* t_lfc1 = tb + (size_t)522 * 512;  // 64
  _Float16* t_lfc2 = tb + (size_t)586 * 512;  // 256

  prep_btiles<<<24, 32, 0, stream>>>(a_whh, t_awhh, 64, 2);
  prep_btiles<<<64, 32, 0, stream>>>(a_fc1w, t_afc1, 64, 2);
  prep_btiles<<<128, 32, 0, stream>>>(a_fc2w, t_afc2, 512, 16);
  prep_btiles<<<12, 32, 0, stream>>>(c_wih, t_cwih, 35, 2);
  prep_btiles<<<6, 32, 0, stream>>>(c_whh, t_cwhh, 32, 1);
  prep_btiles<<<32, 32, 0, stream>>>(c_fc1w, t_cfc1, 32, 1);
  prep_btiles<<<256, 32, 0, stream>>>(c_fc2w, t_cfc2, 512, 16);
  prep_btiles<<<64, 32, 0, stream>>>(l_fc1w, t_lfc1, 35, 2);
  prep_btiles<<<256, 32, 0, stream>>>(l_fc2w, t_lfc2, 512, 16);

  actor_kernel<<<6400, 32, 0, stream>>>(x, a_wih, a_bih, a_bhh, t_awhh, t_afc1,
                                        a_fc1b, t_afc2, a_fc2b, a_fc3w, a_fc3b,
                                        sa);

  gather_p<<<(102400 + 255) / 256, 256, 0, stream>>>(sa, (float*)d_out);

  critic_kernel<<<64, 32, 0, stream>>>(
      x, sa, t_cwih, t_cwhh, c_bih, c_bhh, t_cfc1, c_fc1b, t_cfc2, c_fc2b,
      c_fc3w, c_fc3b, t_lfc1, l_fc1b, t_lfc2, l_fc2b, l_fc3w, l_fc3b,
      (float*)d_out + 102400);
}
